// LSTMClassifier_14121852469556
// MI455X (gfx1250) — compile-verified, matrix operand order checked by reading, not run on hardware
//
#include <hip/hip_runtime.h>
#include <math.h>

// ---------------------------------------------------------------------------
// MI455X (gfx1250) biLSTM. wave32. WMMA bf16 16x16x32, f32 accumulate.
//   - k_cast_bf16   : f32 -> bf16 conversions (weights, x)
//   - k_xproj<DIN>  : xp = inp @ wih.T + b, K-outer tiling, 128-VGPR acc,
//                     output in WMMA C-fragment-major layout
//   - k_lstm_layer  : persistent recurrence; whh slice pre-swizzled into 64KB
//                     fragment-major LDS (reused for all T steps); h via L2
//                     ping-pong; device-scope spin barrier; xp prefetch
//   - k_fc          : final dense on out[:, T-1, :]
// ---------------------------------------------------------------------------

typedef __attribute__((ext_vector_type(16))) __bf16        v16bf;
typedef __attribute__((ext_vector_type(8)))  float         v8f;
typedef __attribute__((ext_vector_type(4)))  unsigned int  v4u;

struct bfpair { v4u lo, hi; };
struct b8arr  { __bf16 e[8]; };     // 16 bytes: one C-fragment per lane (bf16)

__device__ __forceinline__ v16bf make_v16bf(v4u lo, v4u hi) {
  bfpair t; t.lo = lo; t.hi = hi;
  return __builtin_bit_cast(v16bf, t);
}

__device__ __forceinline__ __bf16 f2bf(float f) {
  unsigned u = __builtin_bit_cast(unsigned, f);
  unsigned r = u + 0x7FFFu + ((u >> 16) & 1u);           // round-to-nearest-even
  unsigned short s = (unsigned short)(r >> 16);
  return __builtin_bit_cast(__bf16, s);
}
__device__ __forceinline__ float bf2f(__bf16 b) {
  unsigned short s = __builtin_bit_cast(unsigned short, b);
  unsigned u = ((unsigned)s) << 16;
  return __builtin_bit_cast(float, u);
}

__device__ __forceinline__ float sigm(float x)   { return 1.0f / (1.0f + __expf(-x)); }
__device__ __forceinline__ float tanh_f(float x) { float e = __expf(-2.0f * x); return (1.0f - e) / (1.0f + e); }

// Fragment loader per ISA 16-bit A/B layout: one lane holds 8 bf16 at `c0`
// and 8 bf16 at `c0+16` along the contiguous (K) axis. Two 16B loads.
__device__ __forceinline__ v16bf load_frag(const __bf16* __restrict__ base,
                                           size_t row, int rowstride, int c0) {
  const __bf16* p = base + row * (size_t)rowstride + c0;
  v4u lo = *(const v4u*)(p);
  v4u hi = *(const v4u*)(p + 16);
  return make_v16bf(lo, hi);
}

// DIN==16 fragment: K-chunk 0 only; lane half `hi` covers k = hi*8..hi*8+7
// (valid), elements 8..15 cover k = 16+hi*8.. (out of range) -> zero.
__device__ __forceinline__ v16bf load_frag_k16(const __bf16* __restrict__ base,
                                               size_t row, int hi) {
  const __bf16* p = base + row * 16 + hi * 8;
  v4u lo = *(const v4u*)(p);
  v4u z  = {0u, 0u, 0u, 0u};
  return make_v16bf(lo, z);
}

// Fragment-major LDS read: 32 contiguous bytes per (slot, lane).
__device__ __forceinline__ v16bf lds_frag(const __bf16* p) {
  v4u lo = *(const v4u*)(p);
  v4u hi = *(const v4u*)(p + 8);
  return make_v16bf(lo, hi);
}

__device__ __forceinline__ v8f wmma_bf16(v16bf a, v16bf b, v8f c) {
  // (neg_a, A, neg_b, B, c_mod, C, reuse_a, reuse_b)
  return __builtin_amdgcn_wmma_f32_16x16x32_bf16(false, a, false, b, (short)0, c, false, false);
}

// C-fragment-major xp layout: [t][ntile(64)][mtile(16)][lane(32)][v(8)] bf16.
__device__ __forceinline__ size_t frag_off(int t, int nt, int mt, int lane) {
  return ((((size_t)t * 64 + nt) * 16 + mt) * 32 + lane) * 8;
}

// ---------------------------------------------------------------------------
__global__ void k_cast_bf16(const float* __restrict__ src, __bf16* __restrict__ dst,
                            long long n) {
  long long i = (long long)blockIdx.x * blockDim.x + threadIdx.x;
  long long stride = (long long)gridDim.x * blockDim.x;
  for (; i < n; i += stride) dst[i] = f2bf(src[i]);
}

__global__ void k_zero32(unsigned* __restrict__ p, long long n) {
  long long i = (long long)blockIdx.x * blockDim.x + threadIdx.x;
  long long stride = (long long)gridDim.x * blockDim.x;
  for (; i < n; i += stride) p[i] = 0u;
}

// ---------------------------------------------------------------------------
// xp[(t, n, b)] = sum_k inp[b*T+t][k] * w[n][k] + bias[n],  n in [0,1024).
// Grid: blockIdx.x = ((t*4)+bgroup)*2 + nhalf; block = 256 threads (8 waves).
// Block owns 4 M-tiles (64 batches at fixed t) and 32 of 64 N-tiles; each
// wave owns 4 N-tiles -> acc = 4x4 tiles = 128 VGPRs (no spill).
template <int DIN>
__global__ __launch_bounds__(256)
void k_xproj(const __bf16* __restrict__ inp,   // [B*T][DIN] rows b*T+t
             const __bf16* __restrict__ w,     // [1024][DIN]
             const float*  __restrict__ bias,  // [1024]
             __bf16*       __restrict__ xp,    // fragment-major
             int T) {
  const int lane = threadIdx.x & 31;
  const int wave = threadIdx.x >> 5;            // 0..7
  const int lr = lane & 15, hi = lane >> 4;
  const int nhalf  = blockIdx.x & 1;
  const int mt0    = ((blockIdx.x >> 1) & 3) * 4;   // 4 of 16 m-tiles
  const int t      = blockIdx.x >> 3;

  v8f acc[4][4];                                 // [j (ntile)][mt]
#pragma unroll
  for (int j = 0; j < 4; ++j) {
    const float bv = bias[(nhalf * 32 + wave + j * 8) * 16 + lr];
#pragma unroll
    for (int mt = 0; mt < 4; ++mt)
#pragma unroll
      for (int v = 0; v < 8; ++v) acc[j][mt][v] = bv;
  }

  if constexpr (DIN == 16) {
    v16bf afr[4];
#pragma unroll
    for (int mt = 0; mt < 4; ++mt) {
      const size_t b = (size_t)(mt0 + mt) * 16 + lr;      // batch row
      afr[mt] = load_frag_k16(inp, b * (size_t)T + t, hi);
    }
#pragma unroll
    for (int j = 0; j < 4; ++j) {
      const v16bf bf =
          load_frag_k16(w, (size_t)(nhalf * 32 + wave + j * 8) * 16 + lr, hi);
#pragma unroll
      for (int mt = 0; mt < 4; ++mt)
        acc[j][mt] = wmma_bf16(afr[mt], bf, acc[j][mt]);
    }
  } else {
    const int NKC = DIN / 32;
#pragma unroll 4
    for (int kc = 0; kc < NKC; ++kc) {
      const int c0 = kc * 32 + hi * 8;
      v16bf afr[4];
#pragma unroll
      for (int mt = 0; mt < 4; ++mt) {
        const size_t b = (size_t)(mt0 + mt) * 16 + lr;
        afr[mt] = load_frag(inp, b * (size_t)T + t, DIN, c0);
      }
#pragma unroll
      for (int j = 0; j < 4; ++j) {
        const v16bf bf =
            load_frag(w, (size_t)(nhalf * 32 + wave + j * 8) * 16 + lr, DIN, c0);
#pragma unroll
        for (int mt = 0; mt < 4; ++mt)
          acc[j][mt] = wmma_bf16(afr[mt], bf, acc[j][mt]);
      }
    }
  }

  // Store accumulators fragment-major: one b128 store per (ntile, mtile).
#pragma unroll
  for (int j = 0; j < 4; ++j)
#pragma unroll
    for (int mt = 0; mt < 4; ++mt) {
      b8arr o;
#pragma unroll
      for (int v = 0; v < 8; ++v) o.e[v] = f2bf(acc[j][mt][v]);
      *(v4u*)(xp + frag_off(t, nhalf * 32 + wave + j * 8, mt0 + mt, lane)) =
          __builtin_bit_cast(v4u, o);
    }
}

// ---------------------------------------------------------------------------
// Persistent recurrence for one layer, both directions.
// grid = 16 blocks: dir = blockIdx.x>>3, col-group (32 hidden cols) = blockIdx.x&7.
// Block = 256 threads (8 waves). The block's whh slice is pre-swizzled into
// 64KB of fragment-major LDS once and reused for all T timesteps.
// Per wave: 16 accumulators (128 VGPRs) + cell state (32) + operands (~40).
// h ping-pongs through L2 with a per-direction spin barrier.
#define NCG 8
__global__ __launch_bounds__(256)
void k_lstm_layer(const __bf16* __restrict__ xp,     // frag-major [2 dir]
                  const __bf16* __restrict__ whh_f,  // [1024][256]
                  const __bf16* __restrict__ whh_r,  // [1024][256]
                  __bf16*       __restrict__ hbuf,   // [2 dir][2 parity][256][256]
                  __bf16*       __restrict__ y,      // [256][T][512]
                  unsigned*     __restrict__ sync,   // [2]
                  int T) {
  extern __shared__ char smem_raw[];
  __bf16* wlds = (__bf16*)smem_raw;     // [n16(2)][g(4)][kc(8)][lane(32)][16]

  const int dir  = blockIdx.x >> 3;
  const int cg   = blockIdx.x & 7;
  const int lane = threadIdx.x & 31;
  const int wave = threadIdx.x >> 5;
  const int lr = lane & 15, hi = lane >> 4;

  const __bf16* whh = dir ? whh_r : whh_f;
  const __bf16* xpd = xp + (size_t)dir * ((size_t)T * 64 * 16 * 32 * 8);
  __bf16* hb = hbuf + (size_t)dir * 2 * 256 * 256;
  unsigned* cnt = sync + dir;

  const int n16       = wave & 1;
  const int ncol_base = cg * 32 + n16 * 16;     // this wave's 16 hidden cols
  const int ntc       = cg * 2 + n16;           // n-tile index within a gate
  const int m0        = wave >> 1;              // m tiles m0, m0+4, m0+8, m0+12

  // ---- Stage whh slice into fragment-major LDS (once; step-invariant). ----
  // Slot s = ((n16*4 + g)*8 + kc)*32 + ln holds the 16 bf16 this lane feeds
  // to WMMA as its B operand: w[g*256 + cg*32 + n16*16 + (ln&15)][kc*32 + (ln>>4)*8 (+16)].
  for (int idxs = threadIdx.x; idxs < 2048; idxs += 256) {
    const int ln  = idxs & 31;
    const int kc  = (idxs >> 5) & 7;
    const int g   = (idxs >> 8) & 3;
    const int nn  = idxs >> 10;
    const size_t row = (size_t)(g * 256 + cg * 32 + nn * 16 + (ln & 15));
    const __bf16* src = whh + row * 256 + kc * 32 + (ln >> 4) * 8;
    __bf16* dst = wlds + (size_t)idxs * 16;
    *(v4u*)(dst)     = *(const v4u*)(src);
    *(v4u*)(dst + 8) = *(const v4u*)(src + 16);
  }
  __syncthreads();

  const __bf16* wl_base = wlds + (size_t)(n16 * 4) * 8 * 32 * 16;

  v8f cst[4];
#pragma unroll
  for (int mt = 0; mt < 4; ++mt)
#pragma unroll
    for (int v = 0; v < 8; ++v) cst[mt][v] = 0.0f;

  for (int s = 0; s < T; ++s) {
    const int t = dir ? (T - 1 - s) : s;
    const __bf16* hr = hb + (size_t)(s & 1) * 256 * 256;        // read parity
    __bf16*       hw = hb + (size_t)((s + 1) & 1) * 256 * 256;  // write parity

    // Prefetch next timestep's xp fragments (global_prefetch_b8).
    if (s + 1 < T) {
      const int tn = dir ? (T - 2 - s) : (s + 1);
#pragma unroll
      for (int mt = 0; mt < 4; ++mt)
#pragma unroll
        for (int g = 0; g < 4; ++g)
          __builtin_prefetch(xpd + frag_off(tn, g * 16 + ntc, m0 + mt * 4, lane), 0, 1);
    }

    // Accumulator init: one b128 fragment load per (mt, gate).
    v8f acc[4][4];                                // [mt][g]
#pragma unroll
    for (int mt = 0; mt < 4; ++mt)
#pragma unroll
      for (int g = 0; g < 4; ++g) {
        const v4u raw =
            *(const v4u*)(xpd + frag_off(t, g * 16 + ntc, m0 + mt * 4, lane));
        const b8arr xa = __builtin_bit_cast(b8arr, raw);
#pragma unroll
        for (int v = 0; v < 8; ++v) acc[mt][g][v] = bf2f(xa.e[v]);
      }

    // K-outer: 4 B-fragments from LDS + 4 A-fragments from L2, 16 WMMAs.
#pragma unroll
    for (int kc = 0; kc < 8; ++kc) {
      v16bf bfr[4];
#pragma unroll
      for (int g = 0; g < 4; ++g)
        bfr[g] = lds_frag(wl_base + (size_t)((g * 8 + kc) * 32 + lane) * 16);
#pragma unroll
      for (int mt = 0; mt < 4; ++mt) {
        const int mbase = (m0 + mt * 4) * 16;
        const v16bf a = load_frag(hr, (size_t)(mbase + lr), 256, kc * 32 + hi * 8);
#pragma unroll
        for (int g = 0; g < 4; ++g)
          acc[mt][g] = wmma_bf16(a, bfr[g], acc[mt][g]);
      }
    }

    // LSTM cell update + h stores.
#pragma unroll
    for (int mt = 0; mt < 4; ++mt) {
      const int mbase = (m0 + mt * 4) * 16;
#pragma unroll
      for (int v = 0; v < 8; ++v) {
        const float iv = sigm(acc[mt][0][v]);
        const float fv = sigm(acc[mt][1][v]);
        const float gv = tanh_f(acc[mt][2][v]);
        const float ov = sigm(acc[mt][3][v]);
        const float cv = fv * cst[mt][v] + iv * gv;
        cst[mt][v] = cv;
        const float hv = ov * tanh_f(cv);
        const __bf16 h16 = f2bf(hv);
        const int brow = mbase + hi * 8 + v;
        const int ncol = ncol_base + lr;
        hw[(size_t)brow * 256 + ncol] = h16;
        y[((size_t)brow * T + t) * 512 + dir * 256 + ncol] = h16;
      }
    }

    // Device-scope barrier among the NCG col-groups of this direction.
    __threadfence();
    __syncthreads();
    if (threadIdx.x == 0) {
      atomicAdd(cnt, 1u);
      const unsigned target = (unsigned)NCG * (unsigned)(s + 1);
      while (atomicAdd(cnt, 0u) < target) __builtin_amdgcn_s_sleep(1);
    }
    __syncthreads();
  }
}

// ---------------------------------------------------------------------------
__global__ void k_fc(const __bf16* __restrict__ y,     // [256][T][512]
                     const float* __restrict__ fcw,    // [512]
                     const float* __restrict__ fcb,    // [1]
                     float* __restrict__ out, int T) {
  const int b = threadIdx.x;
  if (b >= 256) return;
  const __bf16* row = y + ((size_t)b * T + (T - 1)) * 512;
  float acc = 0.0f;
#pragma unroll 8
  for (int j = 0; j < 512; ++j) acc += bf2f(row[j]) * fcw[j];
  out[b] = acc + fcb[0];
}

// ---------------------------------------------------------------------------
extern "C" void kernel_launch(void* const* d_in, const int* in_sizes, int n_in,
                              void* d_out, int out_size, void* d_ws, size_t ws_size,
                              hipStream_t stream) {
  (void)in_sizes; (void)n_in; (void)out_size; (void)ws_size;
  const int B = 256, T = 512;
  const int din[3] = {16, 512, 512};

  const float* x = (const float*)d_in[0];
  const float *wih[3][2], *whh[3][2], *bias[3][2];
  int idx = 1;
  for (int l = 0; l < 3; ++l)
    for (int d = 0; d < 2; ++d) {
      wih[l][d]  = (const float*)d_in[idx++];
      whh[l][d]  = (const float*)d_in[idx++];
      bias[l][d] = (const float*)d_in[idx++];
    }
  const float* fcw = (const float*)d_in[19];
  const float* fcb = (const float*)d_in[20];
  float* out = (float*)d_out;

  // Workspace carve (~800 MB): bf16 weights + x + 2 ping-pong layer outputs +
  // fragment-major xp (both dirs, reused per layer) + h ping-pong + counters.
  char* base = (char*)d_ws;
  size_t off = 0;
  auto carve = [&](size_t bytes) -> char* {
    off = (off + 255) & ~(size_t)255;
    char* p = base + off; off += bytes; return p;
  };

  __bf16* x_bf = (__bf16*)carve((size_t)B * T * 16 * 2);
  __bf16 *wih_bf[3][2], *whh_bf[3][2];
  for (int l = 0; l < 3; ++l)
    for (int d = 0; d < 2; ++d) {
      wih_bf[l][d] = (__bf16*)carve((size_t)1024 * din[l] * 2);
      whh_bf[l][d] = (__bf16*)carve((size_t)1024 * 256 * 2);
    }
  __bf16* yA   = (__bf16*)carve((size_t)B * T * 512 * 2);
  __bf16* yB   = (__bf16*)carve((size_t)B * T * 512 * 2);
  const size_t xp_dir = (size_t)T * 64 * 16 * 32 * 8;     // elems per direction
  __bf16* xpws = (__bf16*)carve(2 * xp_dir * 2);
  __bf16* hbuf = (__bf16*)carve((size_t)2 * 2 * 256 * 256 * 2);
  unsigned* syncc = (unsigned*)carve(64);

  // --- casts ---
  k_cast_bf16<<<512, 256, 0, stream>>>(x, x_bf, (long long)B * T * 16);
  for (int l = 0; l < 3; ++l)
    for (int d = 0; d < 2; ++d) {
      k_cast_bf16<<<512, 256, 0, stream>>>(wih[l][d], wih_bf[l][d], (long long)1024 * din[l]);
      k_cast_bf16<<<512, 256, 0, stream>>>(whh[l][d], whh_bf[l][d], (long long)1024 * 256);
    }

  // --- layers ---
  const __bf16* layer_in = x_bf;
  __bf16* layer_out = yA;
  for (int l = 0; l < 3; ++l) {
    for (int d = 0; d < 2; ++d) {
      __bf16* xpd = xpws + (size_t)d * xp_dir;
      if (l == 0)
        k_xproj<16><<<T * 8, 256, 0, stream>>>(layer_in, wih_bf[l][d], bias[l][d], xpd, T);
      else
        k_xproj<512><<<T * 8, 256, 0, stream>>>(layer_in, wih_bf[l][d], bias[l][d], xpd, T);
    }
    // zero h ping-pong + sync counters for this layer
    k_zero32<<<256, 256, 0, stream>>>((unsigned*)hbuf,
                                      (long long)(2 * 2 * 256 * 256 * 2 / 4));
    k_zero32<<<1, 64, 0, stream>>>(syncc, 16);

    k_lstm_layer<<<16, 256, 65536, stream>>>(xpws, whh_bf[l][0], whh_bf[l][1],
                                             hbuf, layer_out, syncc, T);
    layer_in = layer_out;
    layer_out = (layer_out == yA) ? yB : yA;
  }

  k_fc<<<1, 256, 0, stream>>>(layer_in, fcw, fcb, out, T);
}
